// MultiHeadLSRAttentionTriton_63831803953612
// MI455X (gfx1250) — matrix operand order repaired and checked by reading, not compiled, not run on hardware
//
#include <hip/hip_runtime.h>

// ---------------------------------------------------------------- constants
#define B_  2
#define T_  2048
#define DM  1024
#define H_  16
#define DH  64
#define R_  32
#define M_  (B_ * T_)   // 4096 rows of x / q / k / v / y

typedef __attribute__((ext_vector_type(16))) __bf16 bf16x16;
typedef __attribute__((ext_vector_type(8)))  float  f32x8;

union FragU { bf16x16 v; uint4 q[2]; };

static __device__ __forceinline__ int lane_id() { return threadIdx.x & 31; }

// ---- A fragment (16 rows x 32 K, row-major src, leading dim ld) -----------
// Lane layout: lane<16 -> row=lane, K=0..15 ;
// lane>=16 -> row=lane-16, K=16..31.  Two contiguous 16B chunks / lane.
static __device__ __forceinline__ bf16x16 load_a_bf16(const __bf16* t, int ld) {
    int l  = lane_id();
    int kb = (l >> 4) << 4;
    const __bf16* p = t + (size_t)(l & 15) * ld + kb;
    FragU f;
    f.q[0] = *(const uint4*)p;          // K = kb .. kb+7
    f.q[1] = *(const uint4*)(p + 8);    // K = kb+8 .. kb+15
    return f.v;
}

// ---- B fragment from column-major source: element (k,col) at t[col*ld + k].
// Lane layout: lane<16 -> col=lane, K=0..15 ; lane>=16 -> col=lane-16, K=16..31.
static __device__ __forceinline__ bf16x16 load_b_trans_bf16(const __bf16* t, int ld) {
    int l  = lane_id();
    int kb = (l >> 4) << 4;
    const __bf16* p = t + (size_t)(l & 15) * ld + kb;
    FragU f;
    f.q[0] = *(const uint4*)p;          // K = kb .. kb+7
    f.q[1] = *(const uint4*)(p + 8);    // K = kb+8 .. kb+15
    return f.v;
}

// ---- B fragment from fragment-major packed storage (lane*16 contiguous) ---
static __device__ __forceinline__ bf16x16 load_frag_packed(const __bf16* base) {
    const uint4* p = (const uint4*)(base + (size_t)lane_id() * 16);
    FragU f;
    f.q[0] = p[0];
    f.q[1] = p[1];
    return f.v;
}

#define WMMA_BF16(A, Bf, C) \
    __builtin_amdgcn_wmma_f32_16x16x32_bf16(false, (A), false, (Bf), (short)0, (C), false, false)

// ---- async global -> LDS (16B per lane), ASYNCcnt-tracked -----------------
static __device__ __forceinline__ void async_copy16(const void* gptr, void* lptr) {
    unsigned lo = (unsigned)(uintptr_t)lptr;                 // low 32b == LDS offset
    unsigned long long ga = (unsigned long long)(uintptr_t)gptr;
    asm volatile("global_load_async_to_lds_b128 %0, %1, off"
                 :: "v"(lo), "v"(ga) : "memory");
}
static __device__ __forceinline__ void wait_async0() {
    asm volatile("s_wait_asynccnt 0x0" ::: "memory");
}

// ---------------------------------------------------------------- x cast (vec4)
__global__ void cast_f32_bf16_v4(const float* __restrict__ in,
                                 __bf16* __restrict__ out, int n4) {
    int i = blockIdx.x * blockDim.x + threadIdx.x;
    if (i >= n4) return;
    float4 v = ((const float4*)in)[i];
    union { __bf16 h[4]; uint2 u; } r;
    r.h[0] = (__bf16)v.x; r.h[1] = (__bf16)v.y;
    r.h[2] = (__bf16)v.z; r.h[3] = (__bf16)v.w;
    ((uint2*)out)[i] = r.u;
}

// ---------------------------------------------------------------- weight pack
// f32 W[K,N] -> bf16 fragment-major: frag f = kt*(N/16)+nt, lane l, 16 elems.
// Element i of lane l in frag (kt,nt):  K = kt*32 + ((l>>4)<<4) + i,
//                                       col = nt*16 + (l&15).
__global__ __launch_bounds__(256) void pack_w_bf16(const float* __restrict__ W,
                                                   __bf16* __restrict__ out,
                                                   int N, int K) {
    int gid = blockIdx.x * blockDim.x + threadIdx.x;
    int l   = gid & 31;
    int fr  = gid >> 5;
    int ntiles = N >> 4;
    int kt = fr / ntiles, nt = fr % ntiles;
    int col   = nt * 16 + (l & 15);
    int kbase = kt * 32 + ((l >> 4) << 4);
    FragU f;
#pragma unroll
    for (int i = 0; i < 16; ++i)
        f.v[i] = (__bf16)W[(size_t)(kbase + i) * N + col];
    uint4* o = (uint4*)(out + (size_t)gid * 16);
    o[0] = f.q[0]; o[1] = f.q[1];
}

// ---------------------------------------------------------------- V pack
// v[B,T,DM] bf16 -> fragment-major [bh][T/32][4 ntile][lane][16]
__global__ __launch_bounds__(256) void pack_v_bf16(const __bf16* __restrict__ v,
                                                   __bf16* __restrict__ out) {
    int gid = blockIdx.x * blockDim.x + threadIdx.x;
    int l  = gid & 31;
    int n  = (gid >> 5) & 3;
    int st = (gid >> 7) & (T_ / 32 - 1);
    int bh = gid >> 13;                       // (T_/32)=64 -> 6 bits + 7
    int bi = bh >> 4, h = bh & 15;
    int col = n * 16 + (l & 15);
    int kb  = (l >> 4) << 4;
    const __bf16* src = v + ((size_t)bi * T_ + st * 32 + kb) * DM + h * DH + col;
    FragU f;
#pragma unroll
    for (int i = 0; i < 16; ++i) f.v[i] = src[(size_t)i * DM];
    uint4* o = (uint4*)(out + (size_t)gid * 16);
    o[0] = f.q[0]; o[1] = f.q[1];
}

// ---------------------------------------------------------------- tiled WMMA GEMM
// C[M,N] = A[M,K] * Bp + bias.  Bp is fragment-major packed.  Block: 256 thr
// = 8 waves (4x2), tile 128x128, K step 32, double-buffered async LDS staging.
template <bool OUTF32>
__global__ __launch_bounds__(256) void gemm_bf16_wmma(
    const __bf16* __restrict__ A, const __bf16* __restrict__ Bp,
    const float* __restrict__ bias, void* __restrict__ Cout,
    int M, int N, int K)
{
    __shared__ __align__(16) __bf16 As[2][128 * 32];   // 2 x 8 KB
    __shared__ __align__(16) __bf16 Bs[2][8 * 512];    // 2 x 8 KB (8 frags)

    const int tid  = threadIdx.x;
    const int wave = tid >> 5, lane = tid & 31;
    const int wm = wave >> 1, wn = wave & 1;           // 4x2 waves -> 32x64 each
    const int row0 = blockIdx.x * 128;
    const int col0 = blockIdx.y * 128;
    const int ntiles = N >> 4;

    // stage one K-slab: A 128x32 row-major, B 8 packed fragments (contiguous 8KB)
    auto stage = [&](int buf, int k0) {
#pragma unroll
        for (int j = 0; j < 2; ++j) {                  // A: 2 x b128 per thread
            int u = tid + j * 256;
            int r = u >> 2, c = (u & 3) << 3;
            async_copy16(&A[(size_t)(row0 + r) * K + k0 + c], &As[buf][r * 32 + c]);
        }
        const __bf16* bsrc = Bp + ((size_t)(k0 >> 5) * ntiles + (col0 >> 4)) * 512;
#pragma unroll
        for (int j = 0; j < 2; ++j) {                  // B: 2 x b128 per thread
            int u = (tid + j * 256) << 3;              // element offset (8 per b128)
            async_copy16(bsrc + u, &Bs[buf][0] + u);
        }
    };

    f32x8 acc[2][4] = {};
    const int niter = K >> 5;

    stage(0, 0);
    wait_async0();
    __syncthreads();

    for (int it = 0; it < niter; ++it) {
        int cur = it & 1;
        if (it + 1 < niter) stage(cur ^ 1, (it + 1) << 5);

        bf16x16 a0 = load_a_bf16(&As[cur][(wm * 32 +  0) * 32], 32);
        bf16x16 a1 = load_a_bf16(&As[cur][(wm * 32 + 16) * 32], 32);
#pragma unroll
        for (int n = 0; n < 4; ++n) {
            bf16x16 bf = load_frag_packed(&Bs[cur][(wn * 4 + n) * 512]);
            acc[0][n] = WMMA_BF16(a0, bf, acc[0][n]);
            acc[1][n] = WMMA_BF16(a1, bf, acc[1][n]);
        }
        wait_async0();
        __syncthreads();
    }

    const int hi = (lane >> 4) << 3, cl = lane & 15;
#pragma unroll
    for (int i2 = 0; i2 < 2; ++i2)
#pragma unroll
        for (int n = 0; n < 4; ++n)
#pragma unroll
            for (int r = 0; r < 8; ++r) {
                int row = row0 + wm * 32 + i2 * 16 + r + hi;
                int col = col0 + wn * 64 + n * 16 + cl;
                float val = acc[i2][n][r] + (bias ? bias[col] : 0.0f);
                if (OUTF32) ((float*)Cout)[(size_t)row * N + col] = val;
                else        ((__bf16*)Cout)[(size_t)row * N + col] = (__bf16)val;
            }
}

// ---------------------------------------------------------------- low-rank projection
__global__ __launch_bounds__(256) void lowrank_proj(
    const __bf16* __restrict__ qk,   // [M_, DM] bf16 (head-interleaved)
    const float*  __restrict__ Wl,   // [H, DH, R] f32
    const float*  __restrict__ core, // [H, R] or nullptr
    float scale,
    __bf16* __restrict__ out)        // [B, H, T, R] bf16
{
    const int wave = threadIdx.x >> 5, lane = threadIdx.x & 31;
    const int h    = blockIdx.y;
    const int row0 = (blockIdx.x * 8 + wave) * 16;
    const int cl = lane & 15, kb16 = (lane >> 4) << 4, hi = (lane >> 4) << 3;

    f32x8 acc[2] = {};
#pragma unroll
    for (int k0 = 0; k0 < DH; k0 += 32) {
        bf16x16 a = load_a_bf16(qk + (size_t)row0 * DM + h * DH + k0, DM);
#pragma unroll
        for (int n = 0; n < 2; ++n) {
            bf16x16 bfr;
            const float* p = Wl + ((size_t)h * DH + k0) * R_ + n * 16 + cl;
#pragma unroll
            for (int i = 0; i < 16; ++i) bfr[i] = (__bf16)p[(size_t)(kb16 + i) * R_];
            acc[n] = WMMA_BF16(a, bfr, acc[n]);
        }
    }
#pragma unroll
    for (int n = 0; n < 2; ++n) {
        float cs = core ? (scale * core[h * R_ + n * 16 + cl]) : scale;
#pragma unroll
        for (int r = 0; r < 8; ++r) {
            int row = row0 + r + hi;               // global b*T+t
            int bi = row >> 11, t = row & (T_ - 1);
            out[(((size_t)bi * H_ + h) * T_ + t) * R_ + n * 16 + cl] =
                (__bf16)(acc[n][r] * cs);
        }
    }
}

// ---------------------------------------------------------------- flash causal LSR attention
__global__ __launch_bounds__(256) void flash_lsr_attn(
    const __bf16* __restrict__ qlr,   // [B,H,T,R]
    const __bf16* __restrict__ klr,   // [B,H,T,R]
    const __bf16* __restrict__ vpack, // [bh][T/32][4][32][16]
    __bf16* __restrict__ y)           // [B,T,DM]
{
    __shared__ __align__(16) __bf16 pstage[8][16 * 32];  // per-wave P staging

    const int wave = threadIdx.x >> 5, lane = threadIdx.x & 31;
    const int bh = blockIdx.y, bi = bh >> 4, h = bh & 15;
    const int q0 = blockIdx.x * 128 + wave * 16;
    const int hi = (lane >> 4) << 3, cl = lane & 15;

    const __bf16* qb = qlr + ((size_t)bh * T_ + q0) * R_;
    const __bf16* kb = klr + (size_t)bh * T_ * R_;
    const __bf16* vp = vpack + (size_t)bh * (T_ / 32) * 4 * 512;
    __bf16* st = pstage[wave];

    bf16x16 qa = load_a_bf16(qb, R_);       // one fragment: K = R = 32

    float mrow[8], lrow[8];
#pragma unroll
    for (int r = 0; r < 8; ++r) { mrow[r] = -3.0e38f; lrow[r] = 0.0f; }
    f32x8 o[4] = {};
    const f32x8 zc = {};

    for (int s0 = 0; s0 < q0 + 16; s0 += 32) {
        bf16x16 kf0 = load_b_trans_bf16(kb + (size_t)s0 * R_, R_);
        bf16x16 kf1 = load_b_trans_bf16(kb + (size_t)(s0 + 16) * R_, R_);
        f32x8 c0 = WMMA_BF16(qa, kf0, zc);
        f32x8 c1 = WMMA_BF16(qa, kf1, zc);

#pragma unroll
        for (int r = 0; r < 8; ++r) {
            int row = q0 + r + hi;
            float s0v = (s0 + cl      <= row) ? c0[r] : -3.0e38f;
            float s1v = (s0 + 16 + cl <= row) ? c1[r] : -3.0e38f;
            float mx = fmaxf(s0v, s1v);
            mx = fmaxf(mx, __shfl_xor(mx, 1, 32));
            mx = fmaxf(mx, __shfl_xor(mx, 2, 32));
            mx = fmaxf(mx, __shfl_xor(mx, 4, 32));
            mx = fmaxf(mx, __shfl_xor(mx, 8, 32));
            float mn = fmaxf(mrow[r], mx);
            float fs = __expf(mrow[r] - mn);
            float e0 = __expf(s0v - mn);
            float e1 = __expf(s1v - mn);
            float rs = e0 + e1;
            rs += __shfl_xor(rs, 1, 32);
            rs += __shfl_xor(rs, 2, 32);
            rs += __shfl_xor(rs, 4, 32);
            rs += __shfl_xor(rs, 8, 32);
            lrow[r] = lrow[r] * fs + rs;
            mrow[r] = mn;
            o[0][r] *= fs; o[1][r] *= fs; o[2][r] *= fs; o[3][r] *= fs;
            st[(r + hi) * 32 + cl]      = (__bf16)e0;
            st[(r + hi) * 32 + 16 + cl] = (__bf16)e1;
        }
        asm volatile("s_wait_dscnt 0x0" ::: "memory");  // wave-local LDS RAW fence

        bf16x16 pa = load_a_bf16(st, 32);               // P in A layout
        const __bf16* vbase = vp + ((size_t)(s0 >> 5) * 4) * 512;
#pragma unroll
        for (int n = 0; n < 4; ++n) {
            bf16x16 vf = load_frag_packed(vbase + (size_t)n * 512);
            o[n] = WMMA_BF16(pa, vf, o[n]);
        }
    }

    __bf16* yo = y + ((size_t)bi * T_ + q0) * DM + h * DH;
#pragma unroll
    for (int n = 0; n < 4; ++n)
#pragma unroll
        for (int r = 0; r < 8; ++r)
            yo[(size_t)(r + hi) * DM + n * 16 + cl] = (__bf16)(o[n][r] / lrow[r]);
}

// ---------------------------------------------------------------- host launcher
extern "C" void kernel_launch(void* const* d_in, const int* in_sizes, int n_in,
                              void* d_out, int out_size, void* d_ws, size_t ws_size,
                              hipStream_t stream) {
    const float* x    = (const float*)d_in[0];
    const float* Wq   = (const float*)d_in[1];
    const float* bq   = (const float*)d_in[2];
    const float* Wk   = (const float*)d_in[3];
    const float* bk   = (const float*)d_in[4];
    const float* Wv   = (const float*)d_in[5];
    const float* bv   = (const float*)d_in[6];
    const float* Wo   = (const float*)d_in[7];
    const float* bo   = (const float*)d_in[8];
    const float* Wql  = (const float*)d_in[9];
    const float* Wkl  = (const float*)d_in[10];
    const float* core = (const float*)d_in[11];
    (void)in_sizes; (void)n_in; (void)out_size; (void)ws_size;

    const size_t XN = (size_t)M_ * DM;            // 4 Mi elems
    const size_t WN = (size_t)DM * DM;            // 1 Mi elems
    const size_t LN = (size_t)B_ * H_ * T_ * R_;  // 2 Mi elems

    size_t off = 0;
    auto alloc = [&](size_t bytes) {
        void* p = (char*)d_ws + off;
        off += (bytes + 255) & ~(size_t)255;
        return p;
    };
    __bf16* xb   = (__bf16*)alloc(XN * 2);
    __bf16* wqp  = (__bf16*)alloc(WN * 2);   // packed weights
    __bf16* wkp  = (__bf16*)alloc(WN * 2);
    __bf16* wvp  = (__bf16*)alloc(WN * 2);
    __bf16* wop  = (__bf16*)alloc(WN * 2);
    __bf16* qb   = (__bf16*)alloc(XN * 2);
    __bf16* kbuf = (__bf16*)alloc(XN * 2);
    __bf16* vbuf = (__bf16*)alloc(XN * 2);
    __bf16* vpk  = (__bf16*)alloc(XN * 2);   // packed V (same elem count)
    __bf16* qlr  = (__bf16*)alloc(LN * 2);
    __bf16* klr  = (__bf16*)alloc(LN * 2);
    __bf16* ybuf = (__bf16*)alloc(XN * 2);

    // 1) cast x, pack weights into fragment-major bf16
    cast_f32_bf16_v4<<<(int)(XN / 4 / 256), 256, 0, stream>>>(x, xb, (int)(XN / 4));
    const int packg = (DM / 32) * (DM / 16) * 32 / 256;   // 256 blocks
    pack_w_bf16<<<packg, 256, 0, stream>>>(Wq, wqp, DM, DM);
    pack_w_bf16<<<packg, 256, 0, stream>>>(Wk, wkp, DM, DM);
    pack_w_bf16<<<packg, 256, 0, stream>>>(Wv, wvp, DM, DM);
    pack_w_bf16<<<packg, 256, 0, stream>>>(Wo, wop, DM, DM);

    // 2) Q/K/V projections (bf16 out)
    dim3 gg(M_ / 128, DM / 128);
    gemm_bf16_wmma<false><<<gg, 256, 0, stream>>>(xb, wqp, bq, qb,   M_, DM, DM);
    gemm_bf16_wmma<false><<<gg, 256, 0, stream>>>(xb, wkp, bk, kbuf, M_, DM, DM);
    gemm_bf16_wmma<false><<<gg, 256, 0, stream>>>(xb, wvp, bv, vbuf, M_, DM, DM);

    // 3) pack V for the attention inner loop
    pack_v_bf16<<<(B_ * H_ * (T_ / 32) * 4 * 32) / 256, 256, 0, stream>>>(vbuf, vpk);

    // 4) low-rank projections (core * 1/sqrt(R) on Q side)
    const float scale = 0.17677669529663687f;   // 1/sqrt(32)
    dim3 gl(M_ / (16 * 8), H_);
    lowrank_proj<<<gl, 256, 0, stream>>>(qb,   Wql, core, scale, qlr);
    lowrank_proj<<<gl, 256, 0, stream>>>(kbuf, Wkl, nullptr, 1.0f, klr);

    // 5) flash causal attention
    dim3 ga(T_ / 128, B_ * H_);
    flash_lsr_attn<<<ga, 256, 0, stream>>>(qlr, klr, vpk, ybuf);

    // 6) output projection (f32 out to d_out, + bo)
    gemm_bf16_wmma<true><<<gg, 256, 0, stream>>>(ybuf, wop, bo, d_out, M_, DM, DM);
}
